// Head_1623497638718
// MI455X (gfx1250) — matrix operand order repaired and checked by reading, not compiled, not run on hardware
//
#include <hip/hip_runtime.h>
#include <hip/hip_bf16.h>
#include <stdint.h>

// ---------------------------------------------------------------------------
// Types for CDNA5 WMMA bf16 path
// ---------------------------------------------------------------------------
typedef __attribute__((ext_vector_type(16))) __bf16 v16bf;
typedef __attribute__((ext_vector_type(8)))  float  v8f;
typedef __attribute__((ext_vector_type(4)))  int    v4i;

union Frag16 {
    uint32_t u[8];
    v16bf    v;
};

// Native f32 -> bf16 converts (compiler emits v_cvt_* instead of manual RNE)
__device__ __forceinline__ uint16_t bf(float f) {
    union { __bf16 h; uint16_t u; } c;
    c.h = (__bf16)f;
    return c.u;
}
__device__ __forceinline__ uint32_t pack2bf(float a, float b) {
    union { __bf16 h[2]; uint32_t u; } c;
    c.h[0] = (__bf16)a;
    c.h[1] = (__bf16)b;
    return c.u;
}

__device__ __forceinline__ v8f wmma_bf16(v16bf a, v16bf b, v8f c) {
    // D = A(16x32 bf16) * B(32x16 bf16) + C(16x16 f32)
    return __builtin_amdgcn_wmma_f32_16x16x32_bf16(
        /*neg_a=*/false, a, /*neg_b=*/false, b,
        /*c_mod=*/(short)0, c, /*reuse_a=*/false, /*reuse_b=*/false);
}

// A fragment (16x32, bf16) from LDS, row-major [m][k], stride in elements.
__device__ __forceinline__ v16bf load_a_frag(const uint16_t* base, int stride, int lane) {
    int m  = lane & 15;
    int hi = lane >> 4;
    Frag16 f;
#pragma unroll
    for (int j = 0; j < 8; ++j) {
        int k = (j < 4) ? (2 * j + 8 * hi) : (16 + 2 * (j - 4) + 8 * hi);
        f.u[j] = *(const uint32_t*)(base + m * stride + k);  // bf16 pair (k,k+1)
    }
    return f.v;
}

// B fragment (32x16, bf16) from LDS stored n-major [n][k], stride in elements.
__device__ __forceinline__ v16bf load_b_frag(const uint16_t* base, int stride, int lane) {
    int n  = lane & 15;
    int hi = lane >> 4;
    Frag16 f;
#pragma unroll
    for (int j = 0; j < 8; ++j) {
        int k = (j < 4) ? (2 * j + 8 * hi) : (16 + 2 * (j - 4) + 8 * hi);
        f.u[j] = *(const uint32_t*)(base + n * stride + k);  // bf16 pair (k,k+1)
    }
    return f.v;
}

// ---------------------------------------------------------------------------
// Constants
// ---------------------------------------------------------------------------
#define BB 512
#define TT 256
#define CC 256
#define HS 64

// CDNA5 async global->LDS path (ASYNCcnt-tracked)
#if defined(__has_builtin)
#if __has_builtin(__builtin_amdgcn_global_load_async_to_lds_b128) && \
    __has_builtin(__builtin_amdgcn_s_wait_asynccnt)
#define HAVE_ASYNC_LDS 1
#endif
#endif

// ---------------------------------------------------------------------------
// Kernel 1: fused QKV projection. out = x[131072,256] @ [Wq|Wk|Wv][256,192]
// q,k stored row-major bf16 [B*T][64]; v stored transposed bf16 [B][64][T].
// Block: 256 threads (8 waves), 128 rows x 192 cols tile, K chunked by 32.
// ---------------------------------------------------------------------------
__global__ __launch_bounds__(256)
void qkv_kernel(const float* __restrict__ x,
                const float* __restrict__ Wq,
                const float* __restrict__ Wk,
                const float* __restrict__ Wv,
                uint16_t* __restrict__ qw,
                uint16_t* __restrict__ kw,
                uint16_t* __restrict__ vtw) {
    __shared__ uint16_t xA[128 * 32];   // [row][k]  A-operand tile, bf16
    __shared__ uint16_t Wt[192 * 32];   // [n][k]    B-operand tile (n-major), bf16

    const int tid  = threadIdx.x;
    const int lane = tid & 31;
    const int wave = tid >> 5;
    const int row0 = blockIdx.x * 128;      // global row in [B*T]

    v8f acc[12];
#pragma unroll
    for (int nt = 0; nt < 12; ++nt) acc[nt] = (v8f)0.0f;

    for (int kc = 0; kc < 8; ++kc) {
        __syncthreads();
        // Stage x chunk: 128 rows x 32 k. float4 reads, packed bf16 pair stores.
#pragma unroll
        for (int i = 0; i < 4; ++i) {
            int f = i * 256 + tid;          // float4 index (1024 total)
            int r = f >> 3;                 // 8 float4 per row
            int c = (f & 7) * 4;
            float4 v = *(const float4*)(x + (size_t)(row0 + r) * CC + kc * 32 + c);
            uint2 p;
            p.x = pack2bf(v.x, v.y);
            p.y = pack2bf(v.z, v.w);
            *(uint2*)(xA + r * 32 + c) = p;
        }
        // Stage Wt chunk: 192 n x 32 k. float4 reads along n, n-major scatter.
#pragma unroll
        for (int i = 0; i < 6; ++i) {
            int f   = i * 256 + tid;        // float4 index (1536 total)
            int mm  = f >> 9;               // 512 float4 per matrix: 0=q 1=k 2=v
            int rem = f & 511;
            int kk  = rem >> 4;             // 0..31
            int n   = (rem & 15) * 4;       // 0..60
            const float* wp = (mm == 0) ? Wq : ((mm == 1) ? Wk : Wv);
            float4 v = *(const float4*)(wp + (size_t)(kc * 32 + kk) * HS + n);
            uint16_t* dst = Wt + (mm * 64 + n) * 32 + kk;
            dst[0]  = bf(v.x);
            dst[32] = bf(v.y);
            dst[64] = bf(v.z);
            dst[96] = bf(v.w);
        }
        __syncthreads();

        // Each wave owns one 16-row M-tile, all 12 N-tiles
        v16bf a = load_a_frag(xA + (wave * 16) * 32, 32, lane);
#pragma unroll
        for (int nt = 0; nt < 12; ++nt) {
            v16bf b = load_b_frag(Wt + nt * 16 * 32, 32, lane);
            acc[nt] = wmma_bf16(a, b, acc[nt]);
        }
    }

    // Epilogue: C/D layout -> bf16 stores
    const int hi = lane >> 4, nn = lane & 15;
    const int gm0 = row0 + wave * 16;
#pragma unroll
    for (int nt = 0; nt < 12; ++nt) {
        int group = nt >> 2;            // 0=q 1=k 2=v
        int n0    = (nt & 3) * 16;
#pragma unroll
        for (int g = 0; g < 8; ++g) {
            int r = gm0 + g + 8 * hi;
            int n = n0 + nn;
            uint16_t bv = bf(acc[nt][g]);
            if (group == 0) {
                qw[(size_t)r * HS + n] = bv;
            } else if (group == 1) {
                kw[(size_t)r * HS + n] = bv;
            } else {
                int b = r >> 8, t = r & 255;
                vtw[((size_t)b * HS + n) * TT + t] = bv;   // v transposed [b][hs][t]
            }
        }
    }
}

// ---------------------------------------------------------------------------
// Kernel 2: fused causal attention per batch.
// LDS: q [256][64] bf16, k [256][64] bf16 (n-major B for S), vT [64][256] bf16
// (n-major B for P@V), per-wave P-staging [16][32] bf16.
// 8 waves, each handles 2 query M-tiles of 16 rows.
// ---------------------------------------------------------------------------
__global__ __launch_bounds__(256)
void attn_kernel(const uint16_t* __restrict__ qw,
                 const uint16_t* __restrict__ kw,
                 const uint16_t* __restrict__ vtw,
                 float* __restrict__ out) {
    extern __shared__ __align__(16) char smem_raw[];
    uint16_t* qs  = (uint16_t*)smem_raw;          // 256*64
    uint16_t* ks  = qs  + TT * HS;                // 256*64
    uint16_t* vts = ks  + TT * HS;                // 64*256
    uint16_t* ps  = vts + HS * TT;                // 8 * 16*32

    const int b    = blockIdx.x;
    const int tid  = threadIdx.x;
    const int lane = tid & 31;
    const int wave = tid >> 5;
    const int hi   = lane >> 4, nn = lane & 15;

    // Stage q, k, vT (each 16384 bf16 = 2048 x 16B)
#if HAVE_ASYNC_LDS
    {
        // builtin expects: (v4i* in global AS, v4i* in LDS AS, imm offset, cpol)
        typedef __attribute__((address_space(1))) v4i* g4p;
        typedef __attribute__((address_space(3))) v4i* l4p;
        g4p gq = (g4p)(qw  + (size_t)b * TT * HS);
        g4p gk = (g4p)(kw  + (size_t)b * TT * HS);
        g4p gv = (g4p)(vtw + (size_t)b * HS * TT);
        l4p lq = (l4p)qs; l4p lk = (l4p)ks; l4p lv = (l4p)vts;
#pragma unroll
        for (int i = 0; i < 8; ++i) {
            int e = i * 256 + tid;
            __builtin_amdgcn_global_load_async_to_lds_b128(gq + e, lq + e, 0, 0);
            __builtin_amdgcn_global_load_async_to_lds_b128(gk + e, lk + e, 0, 0);
            __builtin_amdgcn_global_load_async_to_lds_b128(gv + e, lv + e, 0, 0);
        }
        __builtin_amdgcn_s_wait_asynccnt(0);
    }
#else
    {
        const uint4* gq = (const uint4*)(qw  + (size_t)b * TT * HS);
        const uint4* gk = (const uint4*)(kw  + (size_t)b * TT * HS);
        const uint4* gv = (const uint4*)(vtw + (size_t)b * HS * TT);
        uint4* lq = (uint4*)qs; uint4* lk = (uint4*)ks; uint4* lv = (uint4*)vts;
#pragma unroll
        for (int i = 0; i < 8; ++i) {
            int e = i * 256 + tid;
            lq[e] = gq[e];
            lk[e] = gk[e];
            lv[e] = gv[e];
        }
    }
#endif
    __syncthreads();

    uint16_t* myps = ps + wave * 16 * 32;

    for (int mt = 0; mt < 2; ++mt) {
        const int m0 = (wave * 2 + mt) * 16;      // query tile base row

        // ---- S = q @ k^T (16 rows x 256 cols), K=HS in 2 steps of 32 ----
        v8f s[16];
#pragma unroll
        for (int nt = 0; nt < 16; ++nt) s[nt] = (v8f)0.0f;
#pragma unroll
        for (int kst = 0; kst < 2; ++kst) {
            v16bf a = load_a_frag(qs + m0 * HS + kst * 32, HS, lane);
#pragma unroll
            for (int nt = 0; nt < 16; ++nt) {
                v16bf bb = load_b_frag(ks + (nt * 16) * HS + kst * 32, HS, lane);
                s[nt] = wmma_bf16(a, bb, s[nt]);
            }
        }

        // ---- causal mask + softmax (row = m0 + g + 8*hi, col = nt*16 + nn) ----
        const float NEG = -__builtin_inff();
#pragma unroll
        for (int g = 0; g < 8; ++g) {
            int qrow = m0 + g + 8 * hi;
            float mx = NEG;
#pragma unroll
            for (int nt = 0; nt < 16; ++nt) {
                int kidx = nt * 16 + nn;
                float v = s[nt][g] * 0.0625f;            // * C^-0.5 = 1/16
                v = (kidx <= qrow) ? v : NEG;
                s[nt][g] = v;
                mx = fmaxf(mx, v);
            }
            // reduce max across the 16 lanes holding this row
            mx = fmaxf(mx, __shfl_xor(mx, 1));
            mx = fmaxf(mx, __shfl_xor(mx, 2));
            mx = fmaxf(mx, __shfl_xor(mx, 4));
            mx = fmaxf(mx, __shfl_xor(mx, 8));

            float sum = 0.0f;
#pragma unroll
            for (int nt = 0; nt < 16; ++nt) {
                float p = __expf(s[nt][g] - mx);         // masked -> exp(-inf)=0
                s[nt][g] = p;
                sum += p;
            }
            sum += __shfl_xor(sum, 1);
            sum += __shfl_xor(sum, 2);
            sum += __shfl_xor(sum, 4);
            sum += __shfl_xor(sum, 8);
            float inv = 1.0f / sum;
#pragma unroll
            for (int nt = 0; nt < 16; ++nt) s[nt][g] *= inv;
        }

        // ---- out = P @ V : K = 256 keys in 8 steps of 32 ----
        v8f o[4];
#pragma unroll
        for (int nt = 0; nt < 4; ++nt) o[nt] = (v8f)0.0f;

        for (int kt = 0; kt < 8; ++kt) {
            // Restage P columns [32*kt, 32*kt+32) into A-layout row-major [16][32]
#pragma unroll
            for (int half = 0; half < 2; ++half) {
                int nt = kt * 2 + half;
#pragma unroll
                for (int g = 0; g < 8; ++g) {
                    int m = g + 8 * hi;
                    myps[m * 32 + half * 16 + nn] = bf(s[nt][g]);
                }
            }
            // per-wave private LDS scratch: DS ops are in-order within a wave
            v16bf a = load_a_frag(myps, 32, lane);
#pragma unroll
            for (int nt = 0; nt < 4; ++nt) {
                v16bf bb = load_b_frag(vts + (nt * 16) * TT + kt * 32, TT, lane);
                o[nt] = wmma_bf16(a, bb, o[nt]);
            }
        }

        // ---- store f32 output [b][m][hs] ----
#pragma unroll
        for (int nt = 0; nt < 4; ++nt) {
#pragma unroll
            for (int g = 0; g < 8; ++g) {
                int m = m0 + g + 8 * hi;
                int n = nt * 16 + nn;
                out[((size_t)b * TT + m) * HS + n] = o[nt][g];
            }
        }
    }
}

// ---------------------------------------------------------------------------
// Launch
// ---------------------------------------------------------------------------
extern "C" void kernel_launch(void* const* d_in, const int* in_sizes, int n_in,
                              void* d_out, int out_size, void* d_ws, size_t ws_size,
                              hipStream_t stream) {
    const float* x  = (const float*)d_in[0];
    const float* Wq = (const float*)d_in[1];
    const float* Wk = (const float*)d_in[2];
    const float* Wv = (const float*)d_in[3];
    float* out = (float*)d_out;

    // Workspace partition: q, k (row-major) and vT (transposed), bf16
    const size_t elems = (size_t)BB * TT * HS;   // 8,388,608 each
    uint16_t* qw  = (uint16_t*)d_ws;
    uint16_t* kw  = qw + elems;
    uint16_t* vtw = kw + elems;

    dim3 blk(256);
    dim3 grid1((BB * TT) / 128);    // 1024
    qkv_kernel<<<grid1, blk, 0, stream>>>(x, Wq, Wk, Wv, qw, kw, vtw);

    size_t smem = (size_t)(TT * HS + TT * HS + HS * TT + 8 * 16 * 32) * sizeof(uint16_t);
    dim3 grid2(BB);                 // 512
    attn_kernel<<<grid2, blk, smem, stream>>>(qw, kw, vtw, out);
}
